// MurelCell_62225486185098
// MI455X (gfx1250) — matrix operand order repaired and checked by reading, not compiled, not run on hardware
//
#include <hip/hip_runtime.h>
#include <hip/hip_bf16.h>
#include <cstdint>

#define Bsz   64
#define Nobj  36
#define Dd    2048
#define DATT  512
#define BN    (Bsz * Nobj)       // 2304
#define SLAB  256                // K-slab for LDS staging
#define RLD   (SLAB + 8)         // padded LDS row stride (ushorts): 4-bank rotation
#define NSLAB (Dd / SLAB)        // 8

typedef __attribute__((ext_vector_type(16))) __bf16 v16bf;
typedef __attribute__((ext_vector_type(8)))  __bf16 v8bf;
typedef __attribute__((ext_vector_type(8)))  float  v8f;
typedef __attribute__((ext_vector_type(4)))  int    v4i;

typedef __attribute__((address_space(1))) v4i* gptr_v4i;
typedef __attribute__((address_space(3))) v4i* lptr_v4i;

union FragBF {
    v16bf v;
    v8bf  h[2];
    unsigned short u[16];
};

__device__ __forceinline__ unsigned short f2bf(float f) {
    unsigned int u = __float_as_uint(f);
    unsigned int r = (u + 0x7FFFu + ((u >> 16) & 1u)) >> 16;  // RNE
    return (unsigned short)r;
}

// 16B global -> LDS async copy (ASYNCcnt-tracked, no VGPR data path)
__device__ __forceinline__ void async_b128(void* lds, const void* g) {
    __builtin_amdgcn_global_load_async_to_lds_b128(
        (gptr_v4i)(void*)g, (lptr_v4i)lds, 0, 0);
}
__device__ __forceinline__ void wait_async0() {
    asm volatile("s_wait_asynccnt 0" ::: "memory");
}

__device__ __forceinline__ v8f wmma_bf16(const FragBF& a, const FragBF& b, v8f c) {
    return __builtin_amdgcn_wmma_f32_16x16x32_bf16(false, a.v, false, b.v, (short)0, c, false, false);
}

// ---------------- prep: fp32 -> bf16 (optionally transposed) ----------------
__global__ void k_conv_bf16(const float* __restrict__ s, unsigned short* __restrict__ d, int n) {
    int t = blockIdx.x * blockDim.x + threadIdx.x;
    if (t < n) d[t] = f2bf(s[t]);
}

// s: [K,N] row-major  ->  d: [N,K] row-major (bf16)
__global__ void k_convT_bf16(const float* __restrict__ s, unsigned short* __restrict__ d, int K, int N) {
    int t = blockIdx.x * blockDim.x + threadIdx.x;
    if (t < K * N) {
        int n = t / K, k = t - n * K;
        d[t] = f2bf(s[(size_t)k * N + n]);
    }
}

// ---------------- fused = (q@Wq) * (o@Wo) -----------------------------------
// Block: 128 thr (4 waves), one 16-row M tile; A(q,o) slabs async-staged in LDS
// (double buffered); each wave owns 4 N-tiles (A-fragment reuse x4).
__global__ __launch_bounds__(128)
void k_fused_gemm(const unsigned short* __restrict__ qB,
                  const unsigned short* __restrict__ oB,
                  const unsigned short* __restrict__ WqT,
                  const unsigned short* __restrict__ WoT,
                  float* __restrict__ fusedF,
                  unsigned short* __restrict__ fusedB) {
    __shared__ unsigned short qS[2][16 * RLD];
    __shared__ unsigned short oS[2][16 * RLD];
    const int tid  = threadIdx.x;
    const int lane = tid & 31, wave = tid >> 5;
    const int m16  = lane & 15, half = lane >> 4;
    const int mt   = blockIdx.x;                  // 144 M tiles
    const int nt0  = (blockIdx.y * 4 + wave) * 4; // 4 consecutive N tiles per wave

    auto stage = [&](int buf, int kbase) {
        // 16 rows x 256 bf16 = 512 chunks of 16B across both matrices
        for (int c = tid; c < 16 * 32; c += 128) {
            const int r  = c >> 5;
            const int ch = (c & 31) << 3;  // ushort offset within slab row
            const size_t go = (size_t)(mt * 16 + r) * Dd + kbase + ch;
            const int    lo = r * RLD + ch;
            async_b128(&qS[buf][lo], qB + go);
            async_b128(&oS[buf][lo], oB + go);
        }
    };

    stage(0, 0);
    wait_async0();
    __syncthreads();

    v8f acq[4], aco[4];
#pragma unroll
    for (int j = 0; j < 4; ++j) { v8f z = {}; acq[j] = z; aco[j] = z; }

    for (int slab = 0; slab < NSLAB; ++slab) {
        if (slab + 1 < NSLAB) stage((slab + 1) & 1, (slab + 1) * SLAB);
        const unsigned short* qRow = &qS[slab & 1][m16 * RLD];
        const unsigned short* oRow = &oS[slab & 1][m16 * RLD];
        const int kg = slab * SLAB;
        for (int ks = 0; ks < SLAB; ks += 32) {
            FragBF aq, ao;
            const int ka = ks + half * 16;
            aq.h[0] = *(const v8bf*)(qRow + ka);
            aq.h[1] = *(const v8bf*)(qRow + ka + 8);
            ao.h[0] = *(const v8bf*)(oRow + ka);
            ao.h[1] = *(const v8bf*)(oRow + ka + 8);
            const int kb = kg + ks + half * 16;
#pragma unroll
            for (int j = 0; j < 4; ++j) {
                const int col = (nt0 + j) * 16 + m16;
                FragBF bq, bo;
                bq.v = *(const v16bf*)(WqT + (size_t)col * Dd + kb);
                bo.v = *(const v16bf*)(WoT + (size_t)col * Dd + kb);
                acq[j] = wmma_bf16(aq, bq, acq[j]);
                aco[j] = wmma_bf16(ao, bo, aco[j]);
            }
        }
        wait_async0();
        __syncthreads();
    }
#pragma unroll
    for (int j = 0; j < 4; ++j) {
        const int col = (nt0 + j) * 16 + m16;
#pragma unroll
        for (int r = 0; r < 8; ++r) {
            const int m = mt * 16 + r + 8 * half;
            float f = acq[j][r] * aco[j][r];
            fusedF[(size_t)m * Dd + col] = f;
            fusedB[(size_t)m * Dd + col] = f2bf(f);
        }
    }
}

// ---------------- C1 = A@W1T^T, C2 = A@W2T^T (ra / rb) ----------------------
__global__ __launch_bounds__(128)
void k_dual_gemm(const unsigned short* __restrict__ A,
                 const unsigned short* __restrict__ W1T,
                 const unsigned short* __restrict__ W2T,
                 float* __restrict__ C1, float* __restrict__ C2) {
    __shared__ unsigned short aS[2][16 * RLD];
    const int tid  = threadIdx.x;
    const int lane = tid & 31, wave = tid >> 5;
    const int m16  = lane & 15, half = lane >> 4;
    const int mt   = blockIdx.x;
    const int nt0  = (blockIdx.y * 4 + wave) * 4;

    auto stage = [&](int buf, int kbase) {
        for (int c = tid; c < 16 * 32; c += 128) {
            const int r  = c >> 5;
            const int ch = (c & 31) << 3;
            async_b128(&aS[buf][r * RLD + ch], A + (size_t)(mt * 16 + r) * Dd + kbase + ch);
        }
    };

    stage(0, 0);
    wait_async0();
    __syncthreads();

    v8f ac1[4], ac2[4];
#pragma unroll
    for (int j = 0; j < 4; ++j) { v8f z = {}; ac1[j] = z; ac2[j] = z; }

    for (int slab = 0; slab < NSLAB; ++slab) {
        if (slab + 1 < NSLAB) stage((slab + 1) & 1, (slab + 1) * SLAB);
        const unsigned short* aRow = &aS[slab & 1][m16 * RLD];
        const int kg = slab * SLAB;
        for (int ks = 0; ks < SLAB; ks += 32) {
            FragBF a;
            const int ka = ks + half * 16;
            a.h[0] = *(const v8bf*)(aRow + ka);
            a.h[1] = *(const v8bf*)(aRow + ka + 8);
            const int kb = kg + ks + half * 16;
#pragma unroll
            for (int j = 0; j < 4; ++j) {
                const int col = (nt0 + j) * 16 + m16;
                FragBF b1, b2;
                b1.v = *(const v16bf*)(W1T + (size_t)col * Dd + kb);
                b2.v = *(const v16bf*)(W2T + (size_t)col * Dd + kb);
                ac1[j] = wmma_bf16(a, b1, ac1[j]);
                ac2[j] = wmma_bf16(a, b2, ac2[j]);
            }
        }
        wait_async0();
        __syncthreads();
    }
#pragma unroll
    for (int j = 0; j < 4; ++j) {
        const int col = (nt0 + j) * 16 + m16;
#pragma unroll
        for (int r = 0; r < 8; ++r) {
            const int m = mt * 16 + r + 8 * half;
            C1[(size_t)m * Dd + col] = ac1[j][r];
            C2[(size_t)m * Dd + col] = ac2[j][r];
        }
    }
}

// ---------------- ba = box@Wba, bb = box@Wbb (K=4) --------------------------
__global__ void k_boxproj(const float* __restrict__ boxes, const float* __restrict__ Wba,
                          const float* __restrict__ Wbb, float* __restrict__ ba,
                          float* __restrict__ bb) {
    int t = blockIdx.x * blockDim.x + threadIdx.x;
    if (t >= BN * Dd) return;
    int bn = t >> 11, d = t & (Dd - 1);
    const float* bx = boxes + bn * 4;
    float sa = 0.f, sb = 0.f;
#pragma unroll
    for (int k = 0; k < 4; ++k) {
        float v = bx[k];
        sa += v * Wba[k * Dd + d];
        sb += v * Wbb[k * Dd + d];
    }
    ba[t] = sa;
    bb[t] = sb;
}

// ---------------- per-(b,i): rel slab in LDS -> WMMA @ W0 -> tanh -> @W1
// ---------------- -> softmax(j) -> factored e_hat -> output -----------------
// Wave tile map: 96 tiles = 32 col-tiles x 3 row-tiles; wave w owns col-tiles
// {w, w+16} x row-tiles {0,1,2} so each W0T B-fragment feeds 3 WMMAs.
__global__ __launch_bounds__(512)
void k_relatt(const float* __restrict__ ra, const float* __restrict__ rb,
              const float* __restrict__ ba, const float* __restrict__ bb,
              const unsigned short* __restrict__ W0T,
              const float* __restrict__ b0, const float* __restrict__ W1,
              const float* __restrict__ b1,
              const float* __restrict__ fusedF, const float* __restrict__ obj,
              float* __restrict__ out) {
    __shared__ unsigned short relS[48 * RLD];
    __shared__ float scoresS[48];
    __shared__ float attS[36];

    const int p    = blockIdx.x;        // b*36 + i
    const int base = (p / Nobj) * Nobj;
    const int tid  = threadIdx.x;
    const int lane = tid & 31, wave = tid >> 5;
    const int m16  = lane & 15, half = lane >> 4;

    if (tid < 48) scoresS[tid] = 0.0f;

    const float* raI = ra + (size_t)p * Dd;
    const float* baI = ba + (size_t)p * Dd;

    v8f acc[6];
#pragma unroll
    for (int s = 0; s < 6; ++s) { v8f z = {}; acc[s] = z; }

    for (int slab = 0; slab < NSLAB; ++slab) {
        const int kbase = slab * SLAB;
        __syncthreads();
        // build rel slab: rel[j,k] = ra_i[k]*rb_j[k] + ba_i[k]*bb_j[k]
        for (int e = tid; e < 48 * SLAB; e += 512) {
            int j  = e >> 8;
            int kk = e & (SLAB - 1);
            float v = 0.0f;
            if (j < 36) {
                int k = kbase + kk;
                size_t rj = (size_t)(base + j) * Dd + k;
                v = raI[k] * rb[rj] + baI[k] * bb[rj];
            }
            relS[j * RLD + kk] = f2bf(v);
        }
        __syncthreads();
#pragma unroll
        for (int g = 0; g < 2; ++g) {
            const int nt = wave + 16 * g;
            const int n  = nt * 16 + m16;
            const unsigned short* wRow = W0T + (size_t)n * Dd + kbase;
            for (int ks = 0; ks < SLAB; ks += 32) {
                FragBF bf;
                bf.v = *(const v16bf*)(wRow + ks + half * 16);
                const int ka = ks + half * 16;
#pragma unroll
                for (int rt = 0; rt < 3; ++rt) {
                    const unsigned short* relRow = relS + (rt * 16 + m16) * RLD;
                    FragBF af;
                    af.h[0] = *(const v8bf*)(relRow + ka);
                    af.h[1] = *(const v8bf*)(relRow + ka + 8);
                    acc[g * 3 + rt] = wmma_bf16(af, bf, acc[g * 3 + rt]);
                }
            }
        }
    }
    __syncthreads();
    // scores[j] = sum_n tanh((rel@W0)[j,n] + b0[n]) * W1[n]
#pragma unroll
    for (int s = 0; s < 6; ++s) {
        const int g = s / 3, rt = s % 3;
        const int nt = wave + 16 * g;
        const int n  = nt * 16 + m16;
        const float b0n = b0[n], w1n = W1[n];
#pragma unroll
        for (int r = 0; r < 8; ++r) {
            const int j = rt * 16 + r + 8 * half;
            if (j < 36) atomicAdd(&scoresS[j], tanhf(acc[s][r] + b0n) * w1n);
        }
    }
    __syncthreads();
    if (tid == 0) {
        const float bias = b1[0];
        float mx = -3.4e38f;
        for (int j = 0; j < 36; ++j) {
            float v = scoresS[j] + bias;
            scoresS[j] = v;
            if (v > mx) mx = v;
        }
        float sum = 0.f;
        for (int j = 0; j < 36; ++j) { float e = __expf(scoresS[j] - mx); attS[j] = e; sum += e; }
        const float inv = 1.0f / sum;
        for (int j = 0; j < 36; ++j) attS[j] *= inv;
    }
    __syncthreads();
    // e_hat = ra_i*(sum_j att_j rb_j) + ba_i*(sum_j att_j bb_j)
    for (int d = tid; d < Dd; d += 512) {
        float sr = 0.f, sb = 0.f;
        for (int j = 0; j < 36; ++j) {
            size_t rj = (size_t)(base + j) * Dd + d;
            float a = attS[j];
            sr += a * rb[rj];
            sb += a * bb[rj];
        }
        size_t idx = (size_t)p * Dd + d;
        out[idx] = obj[idx] + fusedF[idx] + raI[d] * sr + baI[d] * sb;
    }
}

// ---------------------------------------------------------------------------
extern "C" void kernel_launch(void* const* d_in, const int* in_sizes, int n_in,
                              void* d_out, int out_size, void* d_ws, size_t ws_size,
                              hipStream_t stream) {
    (void)in_sizes; (void)n_in; (void)out_size; (void)ws_size;
    const float* q    = (const float*)d_in[0];
    const float* obj  = (const float*)d_in[1];
    const float* box  = (const float*)d_in[2];
    const float* Wq   = (const float*)d_in[3];
    const float* Wo   = (const float*)d_in[4];
    const float* Wfa  = (const float*)d_in[5];
    const float* Wfb  = (const float*)d_in[6];
    const float* Wba  = (const float*)d_in[7];
    const float* Wbb  = (const float*)d_in[8];
    const float* W0   = (const float*)d_in[9];
    const float* b0   = (const float*)d_in[10];
    const float* W1   = (const float*)d_in[11];
    const float* b1   = (const float*)d_in[12];
    float* out = (float*)d_out;

    char* ws = (char*)d_ws;
    size_t off = 0;
    auto carve = [&](size_t bytes) { char* p = ws + off; off += (bytes + 255) & ~(size_t)255; return p; };
    unsigned short* qB     = (unsigned short*)carve((size_t)BN * Dd * 2);
    unsigned short* oB     = (unsigned short*)carve((size_t)BN * Dd * 2);
    unsigned short* WqT    = (unsigned short*)carve((size_t)Dd * Dd * 2);
    unsigned short* WoT    = (unsigned short*)carve((size_t)Dd * Dd * 2);
    unsigned short* WfaT   = (unsigned short*)carve((size_t)Dd * Dd * 2);
    unsigned short* WfbT   = (unsigned short*)carve((size_t)Dd * Dd * 2);
    unsigned short* W0T    = (unsigned short*)carve((size_t)DATT * Dd * 2);
    float*          fusedF = (float*)carve((size_t)BN * Dd * 4);
    unsigned short* fusedB = (unsigned short*)carve((size_t)BN * Dd * 2);
    float*          raP    = (float*)carve((size_t)BN * Dd * 4);
    float*          rbP    = (float*)carve((size_t)BN * Dd * 4);
    float*          baP    = (float*)carve((size_t)BN * Dd * 4);
    float*          bbP    = (float*)carve((size_t)BN * Dd * 4);

    const int T   = 256;
    const int nW  = Dd * Dd;
    const int nW0 = Dd * DATT;
    const int nA  = BN * Dd;

    k_convT_bf16<<<(nW  + T - 1) / T, T, 0, stream>>>(Wq,  WqT,  Dd, Dd);
    k_convT_bf16<<<(nW  + T - 1) / T, T, 0, stream>>>(Wo,  WoT,  Dd, Dd);
    k_convT_bf16<<<(nW  + T - 1) / T, T, 0, stream>>>(Wfa, WfaT, Dd, Dd);
    k_convT_bf16<<<(nW  + T - 1) / T, T, 0, stream>>>(Wfb, WfbT, Dd, Dd);
    k_convT_bf16<<<(nW0 + T - 1) / T, T, 0, stream>>>(W0,  W0T,  Dd, DATT);
    k_conv_bf16 <<<(nA  + T - 1) / T, T, 0, stream>>>(q,   qB,   nA);
    k_conv_bf16 <<<(nA  + T - 1) / T, T, 0, stream>>>(obj, oB,   nA);

    // 4 N-tiles per wave -> grid.y = 2048 / (16 * 4 waves * 4 tiles) = 8
    k_fused_gemm<<<dim3(BN / 16, 8), 128, 0, stream>>>(qB, oB, WqT, WoT, fusedF, fusedB);
    k_dual_gemm <<<dim3(BN / 16, 8), 128, 0, stream>>>(fusedB, WfaT, WfbT, raP, rbP);
    k_boxproj   <<<(nA + T - 1) / T, T, 0, stream>>>(box, Wba, Wbb, baP, bbP);
    k_relatt    <<<BN, 512, 0, stream>>>(raP, rbP, baP, bbP, W0T, b0, W1, b1, fusedF, obj, out);
}
